// AttnBlock_2087354106353
// MI455X (gfx1250) — matrix operand order repaired and checked by reading, not compiled, not run on hardware
//
#include <hip/hip_runtime.h>

typedef __attribute__((ext_vector_type(16))) __bf16 v16bf;
typedef __attribute__((ext_vector_type(8)))  float  v8f;
typedef __attribute__((ext_vector_type(4)))  unsigned int u32x4;
typedef __attribute__((ext_vector_type(8)))  unsigned int u32x8;

#define C_DIM 128
#define N_SPATIAL 4096
#define BN 32

union AFrag { v16bf v; unsigned int u[8]; };

// A-frag (16x32 bf16): rows M = lane&15; lane-half selects K pairs {0..3,8..11} vs {4..7,12..15}
static __device__ __forceinline__ v16bf load_a_frag(const __bf16* base, int ldk, int lane, int kbase) {
  const int m = lane & 15;
  const int koff = (lane & 16) ? 4 : 0;
  const unsigned int* p = (const unsigned int*)(base + (size_t)m * ldk + kbase);
  AFrag f;
#pragma unroll
  for (int i = 0; i < 4; ++i) f.u[i] = p[koff + i];
#pragma unroll
  for (int i = 0; i < 4; ++i) f.u[4 + i] = p[8 + koff + i];
  return f.v;
}

// B-frag (32x16 bf16): operand stored [N][K]; lane-half selects K {0..7,16..23} vs {8..15,24..31}
static __device__ __forceinline__ v16bf load_b_frag(const __bf16* base, int ldk, int lane, int kbase) {
  const int n = lane & 15;
  const int koff = (lane & 16) ? 8 : 0;
  const uint4* p = (const uint4*)(base + (size_t)n * ldk + kbase + koff);
  AFrag f;
  uint4 lo = p[0], hi = p[2];
  f.u[0] = lo.x; f.u[1] = lo.y; f.u[2] = lo.z; f.u[3] = lo.w;
  f.u[4] = hi.x; f.u[5] = hi.y; f.u[6] = hi.z; f.u[7] = hi.w;
  return f.v;
}

static __device__ __forceinline__ v8f wmma_bf16(v16bf a, v16bf b, v8f c) {
  return __builtin_amdgcn_wmma_f32_16x16x32_bf16(false, a, false, b, (short)0, c, false, false);
}

static __device__ __forceinline__ float hmax16(float v) {
#pragma unroll
  for (int o = 8; o > 0; o >>= 1) v = fmaxf(v, __shfl_xor(v, o, 32));
  return v;
}
static __device__ __forceinline__ float hsum16(float v) {
#pragma unroll
  for (int o = 8; o > 0; o >>= 1) v += __shfl_xor(v, o, 32);
  return v;
}

// Async global->LDS b128 (ASYNCcnt). LDS byte address = low 32 bits of generic pointer.
static __device__ __forceinline__ void async_load_b128(void* lds, const void* gptr) {
  asm volatile("global_load_async_to_lds_b128 %0, %1, off"
               :: "v"((unsigned int)(size_t)lds), "v"(gptr)
               : "memory");
}

// TDM 1-D copy: n8 elements of 8 bytes, contiguous, global -> LDS (TENSORcnt).
static __device__ __forceinline__ void tdm_load_1d(unsigned int lds_off, const void* gptr, unsigned int n8) {
  unsigned long long ga = (unsigned long long)(size_t)gptr;
  u32x4 g0;
  g0[0] = 1u;                                        // count = 1 valid descriptor
  g0[1] = lds_off;                                   // lds_addr
  g0[2] = (unsigned int)ga;                          // global_addr[31:0]
  g0[3] = (unsigned int)(ga >> 32) | (2u << 30);     // global_addr[56:32] | type=2
  u32x8 g1;
  g1[0] = 3u << 16;                                  // data_size = 8B, mask=0, no flags
  g1[1] = (n8 & 0xffffu) << 16;                      // tensor_dim0[15:0]
  g1[2] = (n8 >> 16) | (1u << 16);                   // tensor_dim0[31:16] | tensor_dim1=1
  g1[3] = (n8 & 0xffffu) << 16;                      // tile_dim0 = n8
  g1[4] = 1u;                                        // tile_dim1 = 1, tile_dim2 = 0
  g1[5] = n8;                                        // tensor_dim0_stride[31:0]
  g1[6] = (n8 & 0xffffu) << 16;                      // stride0[47:32]=0 | tensor_dim1_stride[15:0]
  g1[7] = 0u;                                        // tensor_dim1_stride[47:16]
  asm volatile("tensor_load_to_lds %0, %1" :: "s"(g0), "s"(g1) : "memory");
}

// ---------------- weight fp32 -> bf16 ----------------
__global__ __launch_bounds__(256) void cvt_w_kernel(
    const float* __restrict__ wq, const float* __restrict__ wk,
    const float* __restrict__ wv, const float* __restrict__ wp,
    __bf16* __restrict__ wqb, __bf16* __restrict__ wkb,
    __bf16* __restrict__ wvb, __bf16* __restrict__ wpb) {
  int i = blockIdx.x * 256 + threadIdx.x;          // 64 blocks -> 16384 = 128*128
  wqb[i] = (__bf16)wq[i];
  wkb[i] = (__bf16)wk[i];
  wvb[i] = (__bf16)wv[i];
  wpb[i] = (__bf16)wp[i];
}

// ---------------- GroupNorm: x[B,C,N] -> h_t[B,N,C] bf16 ----------------
__global__ __launch_bounds__(256) void gn_kernel(const float* __restrict__ x,
                                                 const float* __restrict__ gw,
                                                 const float* __restrict__ gb,
                                                 __bf16* __restrict__ h_t) {
  const int b = blockIdx.x >> 5;        // 32 groups/batch, 4 channels/group
  const int g = blockIdx.x & 31;
  const int t = threadIdx.x;
  const float* xp = x + ((size_t)b * C_DIM + g * 4) * N_SPATIAL;  // 16384 contiguous
  float s = 0.f, s2 = 0.f;
  for (int i = t; i < 4 * N_SPATIAL; i += 256) { float v = xp[i]; s += v; s2 += v * v; }
  __shared__ float sh1[256], sh2[256];
  sh1[t] = s; sh2[t] = s2;
  __syncthreads();
  for (int st = 128; st > 0; st >>= 1) {
    if (t < st) { sh1[t] += sh1[t + st]; sh2[t] += sh2[t + st]; }
    __syncthreads();
  }
  const float mean = sh1[0] * (1.f / 16384.f);
  const float var  = sh2[0] * (1.f / 16384.f) - mean * mean;
  const float rinv = rsqrtf(var + 1e-5f);
  for (int i = t; i < 4 * N_SPATIAL; i += 256) {
    int c = g * 4 + (i >> 12);
    int n = i & (N_SPATIAL - 1);
    float v = (xp[i] - mean) * rinv * gw[c] + gb[c];
    h_t[((size_t)b * N_SPATIAL + n) * C_DIM + c] = (__bf16)v;
  }
}

// ---------------- QKV: q/k/v[B,N,C] = h_t[B,N,C] x W[o,c]^T + bias ----------------
__global__ __launch_bounds__(256) void qkv_kernel(
    const __bf16* __restrict__ h_t,
    const __bf16* __restrict__ wqb, const __bf16* __restrict__ wkb, const __bf16* __restrict__ wvb,
    const float* __restrict__ bq, const float* __restrict__ bk, const float* __restrict__ bv,
    __bf16* __restrict__ q_t, __bf16* __restrict__ k_t, __bf16* __restrict__ v_t) {
  const int gw = blockIdx.x * 8 + (threadIdx.x >> 5);   // 4096 tiles total
  const int lane = threadIdx.x & 31;
  const int b = gw >> 11;
  const int rem = gw & 2047;
  const int n0 = (rem >> 3) << 4;
  const int o0 = (rem & 7) << 4;
  const __bf16* arow = h_t + ((size_t)b * N_SPATIAL + n0) * C_DIM;
  v8f cq = {}, ck = {}, cv = {};
#pragma unroll
  for (int kb = 0; kb < 4; ++kb) {
    v16bf a = load_a_frag(arow, C_DIM, lane, kb * 32);
    cq = wmma_bf16(a, load_b_frag(wqb + o0 * C_DIM, C_DIM, lane, kb * 32), cq);
    ck = wmma_bf16(a, load_b_frag(wkb + o0 * C_DIM, C_DIM, lane, kb * 32), ck);
    cv = wmma_bf16(a, load_b_frag(wvb + o0 * C_DIM, C_DIM, lane, kb * 32), cv);
  }
  const int oc = o0 + (lane & 15);
  const float bq_ = bq[oc], bk_ = bk[oc], bv_ = bv[oc];
  const int mh = (lane >> 4) << 3;
#pragma unroll
  for (int j = 0; j < 8; ++j) {
    size_t idx = ((size_t)b * N_SPATIAL + n0 + j + mh) * C_DIM + oc;
    q_t[idx] = (__bf16)(cq[j] + bq_);
    k_t[idx] = (__bf16)(ck[j] + bk_);
    v_t[idx] = (__bf16)(cv[j] + bv_);
  }
}

// ---------------- Flash attention: att[B,N,C] = softmax(q k^T / sqrt(C)) v ----------------
__global__ __launch_bounds__(256) void attn_kernel(const __bf16* __restrict__ q_t,
                                                   const __bf16* __restrict__ k_t,
                                                   const __bf16* __restrict__ v_t,
                                                   __bf16* __restrict__ att) {
  __shared__ __bf16 ksh[BN][C_DIM];      // key-major, filled by TDM: B-frag source for QK^T
  __shared__ __bf16 vtmp[BN][C_DIM];     // key-major, filled by async loads
  __shared__ __bf16 vsh[C_DIM][BN];      // channel-major: B-frag source for PV
  __shared__ __bf16 psh[8][16][BN];      // per-wave P staging (D-layout -> A-layout)
  const int tid = threadIdx.x;
  const int lane = tid & 31;
  const int wave = tid >> 5;
  const int mh = (lane >> 4) << 3;
  const int b = blockIdx.x >> 5;
  const int qbase = (blockIdx.x & 31) * 128;
  const __bf16* kg = k_t + (size_t)b * N_SPATIAL * C_DIM;
  const __bf16* vg = v_t + (size_t)b * N_SPATIAL * C_DIM;
  const __bf16* qrow = q_t + ((size_t)b * N_SPATIAL + qbase + wave * 16) * C_DIM;

  v16bf qf[4];
#pragma unroll
  for (int kb = 0; kb < 4; ++kb) qf[kb] = load_a_frag(qrow, C_DIM, lane, kb * 32);

  v8f acc[8] = {};
  float m[8], l[8];
#pragma unroll
  for (int j = 0; j < 8; ++j) { m[j] = -1e30f; l[j] = 0.0f; }
  const float rscale = 0.0883883476483184f;   // 128^-0.5

#pragma unroll 1
  for (int j0 = 0; j0 < N_SPATIAL; j0 += BN) {
    __syncthreads();   // prior iteration finished consuming ksh/vsh/vtmp

    // K slice (4 rows = 1KB) per wave via Tensor Data Mover
    tdm_load_1d((unsigned int)(size_t)&ksh[wave * 4][0],
                &kg[(size_t)(j0 + wave * 4) * C_DIM],
                (BN / 8) * C_DIM * 2 / 8);

    // V block via async global->LDS, key-major staging
#pragma unroll
    for (int e = tid; e < BN * C_DIM / 8; e += 256) {
      int key = e >> 4;
      int c8 = (e & 15) << 3;
      async_load_b128(&vtmp[key][c8], &vg[(size_t)(j0 + key) * C_DIM + c8]);
    }

    // prefetch next K/V block, one cacheline per lane
    if (j0 + BN < N_SPATIAL) {
      __builtin_prefetch(&kg[(size_t)(j0 + BN) * C_DIM + tid * 32], 0, 3);
      __builtin_prefetch(&vg[(size_t)(j0 + BN) * C_DIM + tid * 32], 0, 3);
    }

    asm volatile("s_wait_asynccnt 0x0" ::: "memory");
    __builtin_amdgcn_s_wait_tensorcnt(0);
    __syncthreads();   // ksh + vtmp now visible block-wide

    // transpose V: vtmp[key][c] -> vsh[c][key]
#pragma unroll
    for (int e = tid; e < BN * C_DIM / 8; e += 256) {
      int key = e >> 4;
      int c8 = (e & 15) << 3;
      uint4 vv = *(const uint4*)&vtmp[key][c8];
      const __bf16* vb = (const __bf16*)&vv;
#pragma unroll
      for (int q2 = 0; q2 < 8; ++q2) vsh[c8 + q2][key] = vb[q2];
    }
    __syncthreads();

    // scores: two 16x16 tiles over 32 keys, K=128
    v8f s0 = {}, s1 = {};
#pragma unroll
    for (int kb = 0; kb < 4; ++kb) {
      v16bf b0 = load_b_frag(&ksh[0][0],  C_DIM, lane, kb * 32);
      v16bf b1 = load_b_frag(&ksh[16][0], C_DIM, lane, kb * 32);
      s0 = wmma_bf16(qf[kb], b0, s0);
      s1 = wmma_bf16(qf[kb], b1, s1);
    }

    // online softmax per row (row = j + 8*lane-half; stats uniform within 16-lane half)
    float r[8];
#pragma unroll
    for (int j = 0; j < 8; ++j) {
      float a0 = s0[j] * rscale;
      float a1 = s1[j] * rscale;
      float mx = hmax16(fmaxf(a0, a1));
      float mn = fmaxf(m[j], mx);
      float p0 = __expf(a0 - mn);
      float p1 = __expf(a1 - mn);
      float rs = hsum16(p0 + p1);
      r[j] = __expf(m[j] - mn);
      l[j] = l[j] * r[j] + rs;
      m[j] = mn;
      psh[wave][j + mh][lane & 15]        = (__bf16)p0;
      psh[wave][j + mh][16 + (lane & 15)] = (__bf16)p1;
    }
    asm volatile("s_wait_dscnt 0" ::: "memory");   // drain same-wave DS stores before A-frag reload
    v16bf pf = load_a_frag(&psh[wave][0][0], BN, lane, 0);
#pragma unroll
    for (int t = 0; t < 8; ++t) {
#pragma unroll
      for (int j = 0; j < 8; ++j) acc[t][j] *= r[j];
      v16bf vf = load_b_frag(&vsh[t * 16][0], BN, lane, 0);
      acc[t] = wmma_bf16(pf, vf, acc[t]);
    }
  }

  float inv[8];
#pragma unroll
  for (int j = 0; j < 8; ++j) inv[j] = 1.0f / l[j];
  __bf16* ao = att + ((size_t)b * N_SPATIAL + qbase + wave * 16) * C_DIM;
#pragma unroll
  for (int t = 0; t < 8; ++t)
#pragma unroll
    for (int j = 0; j < 8; ++j)
      ao[(size_t)(j + mh) * C_DIM + t * 16 + (lane & 15)] = (__bf16)(acc[t][j] * inv[j]);
}

// ---------------- proj + residual: out[B,C,N] = x + att[B,N,C] x Wp[o,c]^T + bp ----------------
__global__ __launch_bounds__(256) void proj_kernel(const __bf16* __restrict__ att,
                                                   const __bf16* __restrict__ wpb,
                                                   const float* __restrict__ bp,
                                                   const float* __restrict__ x,
                                                   float* __restrict__ out) {
  const int gw = blockIdx.x * 8 + (threadIdx.x >> 5);
  const int lane = threadIdx.x & 31;
  const int b = gw >> 11;
  const int rem = gw & 2047;
  const int n0 = (rem >> 3) << 4;
  const int o0 = (rem & 7) << 4;
  const __bf16* arow = att + ((size_t)b * N_SPATIAL + n0) * C_DIM;
  v8f c = {};
#pragma unroll
  for (int kb = 0; kb < 4; ++kb)
    c = wmma_bf16(load_a_frag(arow, C_DIM, lane, kb * 32),
                  load_b_frag(wpb + o0 * C_DIM, C_DIM, lane, kb * 32), c);
  const int o = o0 + (lane & 15);
  const float bb = bp[o];
  const int mh = (lane >> 4) << 3;
#pragma unroll
  for (int j = 0; j < 8; ++j) {
    int n = n0 + j + mh;
    size_t idx = ((size_t)b * C_DIM + o) * N_SPATIAL + n;
    out[idx] = x[idx] + c[j] + bb;
  }
}

extern "C" void kernel_launch(void* const* d_in, const int* in_sizes, int n_in,
                              void* d_out, int out_size, void* d_ws, size_t ws_size,
                              hipStream_t stream) {
  const float* x   = (const float*)d_in[0];
  const float* gnw = (const float*)d_in[1];
  const float* gnb = (const float*)d_in[2];
  const float* wq  = (const float*)d_in[3];
  const float* bq  = (const float*)d_in[4];
  const float* wk  = (const float*)d_in[5];
  const float* bk  = (const float*)d_in[6];
  const float* wv  = (const float*)d_in[7];
  const float* bv  = (const float*)d_in[8];
  const float* wp  = (const float*)d_in[9];
  const float* bp  = (const float*)d_in[10];
  float* out = (float*)d_out;

  __bf16* ws = (__bf16*)d_ws;
  const size_t M = (size_t)2 * N_SPATIAL * C_DIM;   // elements per [B,N,C] tensor
  __bf16* h_t = ws;
  __bf16* q_t = ws + 1 * M;
  __bf16* k_t = ws + 2 * M;
  __bf16* v_t = ws + 3 * M;
  __bf16* att = ws + 4 * M;
  __bf16* wqb = ws + 5 * M;
  __bf16* wkb = wqb + 16384;
  __bf16* wvb = wkb + 16384;
  __bf16* wpb = wvb + 16384;

  hipLaunchKernelGGL(cvt_w_kernel, dim3(64),  dim3(256), 0, stream, wq, wk, wv, wp, wqb, wkb, wvb, wpb);
  hipLaunchKernelGGL(gn_kernel,    dim3(64),  dim3(256), 0, stream, x, gnw, gnb, h_t);
  hipLaunchKernelGGL(qkv_kernel,   dim3(512), dim3(256), 0, stream, h_t, wqb, wkb, wvb, bq, bk, bv, q_t, k_t, v_t);
  hipLaunchKernelGGL(attn_kernel,  dim3(64),  dim3(256), 0, stream, q_t, k_t, v_t, att);
  hipLaunchKernelGGL(proj_kernel,  dim3(512), dim3(256), 0, stream, att, wpb, bp, x, out);
}
